// MambaLayer_48936857371341
// MI455X (gfx1250) — matrix operand order repaired and checked by reading, not compiled, not run on hardware
//
#include <hip/hip_runtime.h>
#include <cstdint>
#include <cstddef>

// ---- problem constants (match reference) ----
#define D_MODEL 1024
#define D_INNER 2048
#define D_CONV  4
#define D_STATE 16
#define DT_RANK 64
#define BSZ     4
#define LSEQ    2048
#define MTOT    (BSZ * LSEQ)      // 8192 rows
#define NSEG    32
#define SEGLEN  (LSEQ / NSEG)     // 64

typedef __bf16 bf16;
typedef __attribute__((ext_vector_type(16))) __bf16 v16bf;
typedef __attribute__((ext_vector_type(8)))  __bf16 v8bf;
typedef __attribute__((ext_vector_type(8)))  float  v8f;

// =====================================================================
// fp32 -> bf16 conversion
// =====================================================================
__global__ void cvt_f32_bf16(const float* __restrict__ in, bf16* __restrict__ out, long n) {
    long i = (long)blockIdx.x * blockDim.x + threadIdx.x;
    if (i < n) out[i] = (bf16)in[i];
}

// Extract dt_lo (first 64 of 96 cols of xdbl) as bf16 for the dt GEMM.
__global__ void extract_dtlo(const float* __restrict__ xdbl, bf16* __restrict__ dtlo, long n) {
    long i = (long)blockIdx.x * blockDim.x + threadIdx.x;
    if (i < n) {
        long m = i >> 6;          // / DT_RANK
        int  r = (int)(i & 63);
        dtlo[i] = (bf16)xdbl[m * 96 + r];
    }
}

// =====================================================================
// WMMA GEMM: C[M,N] (f32) = A[M,K] (bf16, row-major) * W[N,K]^T (bf16, row-major)
// Block: 256 threads = 8 waves (wave32), arranged 2(M) x 4(N).
// Block tile 128x128, wave tile 64x32 -> 4x2 = 8 v_wmma_f32_16x16x32_bf16 / K-step.
// Requires M % 128 == 0 (true: M = 8192) and N % 16 == 0 (true for all four
// GEMMs) so the N guard is tile-uniform: loads are clamped (never OOB),
// stores are predicated per 16-wide tile -> no EXEC churn in the hot loop.
// =====================================================================
__global__ __launch_bounds__(256) void gemm_bf16_wmma(
    const bf16* __restrict__ A, const bf16* __restrict__ W,
    float* __restrict__ C, int M, int N, int K)
{
    const int lane = threadIdx.x & 31;
    const int wave = threadIdx.x >> 5;
    const int wm   = wave & 1;      // 0..1  (M dir)
    const int wn   = wave >> 1;     // 0..3  (N dir)
    const int l16  = lane & 15;
    const int hi   = lane >> 4;     // lane half-select (ISA 16-bit fragment layout)

    const long wave_m = (long)blockIdx.y * 128 + wm * 64;
    const long wave_n = (long)blockIdx.x * 128 + wn * 32;

    v8f zacc = {};
    v8f acc[4][2];
#pragma unroll
    for (int i = 0; i < 4; ++i)
#pragma unroll
        for (int j = 0; j < 2; ++j) acc[i][j] = zacc;

    // Per-lane base pointers.
    // A fragment (16x32 bf16): lane<16 -> row M=l16, K = 0..15; lane>=16 -> K = 16..31 (16 contiguous)
    const bf16* aptr[4];
#pragma unroll
    for (int i = 0; i < 4; ++i)
        aptr[i] = A + (wave_m + 16 * i + l16) * (long)K + hi * 16;

    // B fragment (32x16 bf16): lane<16 -> col N=l16, K=0..15; lane>=16 -> K=16..31 (16 contiguous)
    // Tile-uniform OOB: clamp pointer to row 0 (harmless load), predicate stores only.
    const bf16* bptr[2];
    bool tok[2];
#pragma unroll
    for (int j = 0; j < 2; ++j) {
        long ntile = wave_n + 16 * j;
        tok[j] = (ntile < (long)N);
        long n = tok[j] ? (ntile + l16) : 0;
        bptr[j] = W + n * (long)K + hi * 16;
    }

    union V16 { v16bf v; v8bf h[2]; };

#pragma unroll 2
    for (int k0 = 0; k0 < K; k0 += 32) {
        v16bf a[4], b[2];
#pragma unroll
        for (int i = 0; i < 4; ++i) {
            V16 u;
            u.h[0] = *(const v8bf*)(aptr[i] + k0);        // K block 0..7 (lane half)
            u.h[1] = *(const v8bf*)(aptr[i] + k0 + 8);    // K block 8..15 (lane half)
            a[i] = u.v;
        }
#pragma unroll
        for (int j = 0; j < 2; ++j)
            b[j] = *(const v16bf*)(bptr[j] + k0);         // unconditional (clamped ptr)

        // Stream-prefetch next K tile -> global_prefetch_b8
        if (k0 + 32 < K) {
            __builtin_prefetch((const void*)(aptr[0] + k0 + 32), 0, 0);
            __builtin_prefetch((const void*)(bptr[0] + k0 + 32), 0, 0);
        }

#pragma unroll
        for (int i = 0; i < 4; ++i)
#pragma unroll
            for (int j = 0; j < 2; ++j)
                acc[i][j] = __builtin_amdgcn_wmma_f32_16x16x32_bf16(
                    false, a[i], false, b[j], (short)0, acc[i][j], false, false);
    }

    // C/D layout: lane<16 -> rows 0..7 (VGPR v), lane>=16 -> rows 8..15; col = l16
    const int rb = hi * 8;
#pragma unroll
    for (int i = 0; i < 4; ++i)
#pragma unroll
        for (int j = 0; j < 2; ++j) {
            if (tok[j]) {   // uniform per 16-wide tile (N % 16 == 0)
                long n = wave_n + 16 * j + l16;
                float* cp = C + (wave_m + 16 * i + rb) * (long)N + n;
#pragma unroll
                for (int v = 0; v < 8; ++v) cp[(long)v * N] = acc[i][j][v];
            }
        }
}

// =====================================================================
// Causal depthwise conv (width 4) + SiLU; emits f32 and bf16 copies.
// xz is [M, 2*D_INNER]; xi = cols [0, D_INNER)
// =====================================================================
__global__ void conv_silu_kernel(const float* __restrict__ xz,
                                 const float* __restrict__ conv_w,
                                 const float* __restrict__ conv_b,
                                 float* __restrict__ xc,
                                 bf16*  __restrict__ xc_bf)
{
    long tid = (long)blockIdx.x * blockDim.x + threadIdx.x;
    if (tid >= (long)MTOT * D_INNER) return;
    int  d = (int)(tid % D_INNER);
    long m = tid / D_INNER;
    int  l = (int)(m % LSEQ);

    float acc = conv_b[d];
#pragma unroll
    for (int j = 0; j < D_CONV; ++j) {
        int lj = l - (D_CONV - 1) + j;
        if (lj >= 0)
            acc += conv_w[d * D_CONV + j] * xz[(m - (D_CONV - 1) + j) * (long)(2 * D_INNER) + d];
    }
    float v = acc / (1.0f + __expf(-acc));   // SiLU
    xc[tid]    = v;
    xc_bf[tid] = (bf16)v;
}

// =====================================================================
// Selective scan, parallelized over segments (associative recurrence):
//   h_t[s] = exp(dt_t*A[s]) * h_{t-1}[s] + dt_t*x_t*B_t[s]
// Pass A: per (b,d,seg) compute (P = prod dA, q = local h with h0=0)
// Pass B: per (b,d) sequential 32-segment prefix -> segment start states
// Pass C: replay with correct start state, fuse skip + SiLU(z) gate
// =====================================================================
__device__ __forceinline__ float softplus_f(float x) {
    return x > 20.0f ? x : log1pf(__expf(x));
}

__global__ __launch_bounds__(256) void scan_passA(
    const float* __restrict__ dt_pre, const float* __restrict__ xc,
    const float* __restrict__ xdbl,   const float* __restrict__ b_dt,
    const float* __restrict__ A_log,  float* __restrict__ PQ)
{
    __shared__ float sBC[16][32];   // 16 timesteps x (B[16] | C[16])
    int t = blockIdx.x;
    int dchunk = t & 7;  t >>= 3;
    int seg = t % NSEG;  int b = t / NSEG;
    int d = dchunk * 256 + threadIdx.x;

    float bdt = b_dt[d];
    float Arow[D_STATE];
#pragma unroll
    for (int s = 0; s < D_STATE; ++s) Arow[s] = -__expf(A_log[d * D_STATE + s]);

    float P[D_STATE], h[D_STATE];
#pragma unroll
    for (int s = 0; s < D_STATE; ++s) { P[s] = 1.0f; h[s] = 0.0f; }

    const long mbase = (long)b * LSEQ + (long)seg * SEGLEN;
    for (int t0 = 0; t0 < SEGLEN; t0 += 16) {
        __syncthreads();
        for (int idx = threadIdx.x; idx < 16 * 32; idx += 256) {
            int tt = idx >> 5, c = idx & 31;
            sBC[tt][c] = xdbl[(mbase + t0 + tt) * 96 + DT_RANK + c];
        }
        __syncthreads();
        for (int tt = 0; tt < 16; ++tt) {
            long m   = mbase + t0 + tt;
            float dt = softplus_f(dt_pre[m * (long)D_INNER + d] + bdt);
            float dtx = dt * xc[m * (long)D_INNER + d];
#pragma unroll
            for (int s = 0; s < D_STATE; ++s) {
                float dA = __expf(dt * Arow[s]);
                h[s] = dA * h[s] + dtx * sBC[tt][s];
                P[s] *= dA;
            }
        }
    }
    float* pq = PQ + ((((long)b * D_INNER + d) * NSEG) + seg) * 32;
#pragma unroll
    for (int s = 0; s < D_STATE; ++s) { pq[s] = P[s]; pq[16 + s] = h[s]; }
}

__global__ void scan_passB(const float* __restrict__ PQ, float* __restrict__ Hs) {
    long tid = (long)blockIdx.x * blockDim.x + threadIdx.x;  // b*D_INNER + d
    if (tid >= (long)BSZ * D_INNER) return;
    float h[D_STATE];
#pragma unroll
    for (int s = 0; s < D_STATE; ++s) h[s] = 0.0f;
    const float* pq = PQ + tid * NSEG * 32;
    float*       hs = Hs + tid * NSEG * 16;
    for (int seg = 0; seg < NSEG; ++seg) {
#pragma unroll
        for (int s = 0; s < D_STATE; ++s) hs[seg * 16 + s] = h[s];
#pragma unroll
        for (int s = 0; s < D_STATE; ++s)
            h[s] = pq[seg * 32 + s] * h[s] + pq[seg * 32 + 16 + s];
    }
}

__global__ __launch_bounds__(256) void scan_passC(
    const float* __restrict__ dt_pre, const float* __restrict__ xc,
    const float* __restrict__ xdbl,   const float* __restrict__ xz,
    const float* __restrict__ b_dt,   const float* __restrict__ A_log,
    const float* __restrict__ D_skip, const float* __restrict__ Hs,
    bf16* __restrict__ y_bf)
{
    __shared__ float sBC[16][32];
    int t = blockIdx.x;
    int dchunk = t & 7;  t >>= 3;
    int seg = t % NSEG;  int b = t / NSEG;
    int d = dchunk * 256 + threadIdx.x;

    float bdt = b_dt[d];
    float dsk = D_skip[d];
    float Arow[D_STATE];
#pragma unroll
    for (int s = 0; s < D_STATE; ++s) Arow[s] = -__expf(A_log[d * D_STATE + s]);

    float h[D_STATE];
    const float* hs = Hs + (((long)b * D_INNER + d) * NSEG + seg) * 16;
#pragma unroll
    for (int s = 0; s < D_STATE; ++s) h[s] = hs[s];

    const long mbase = (long)b * LSEQ + (long)seg * SEGLEN;
    for (int t0 = 0; t0 < SEGLEN; t0 += 16) {
        __syncthreads();
        for (int idx = threadIdx.x; idx < 16 * 32; idx += 256) {
            int tt = idx >> 5, c = idx & 31;
            sBC[tt][c] = xdbl[(mbase + t0 + tt) * 96 + DT_RANK + c];
        }
        __syncthreads();
        for (int tt = 0; tt < 16; ++tt) {
            long m   = mbase + t0 + tt;
            float xv = xc[m * (long)D_INNER + d];
            float dt = softplus_f(dt_pre[m * (long)D_INNER + d] + bdt);
            float dtx = dt * xv;
            float y = 0.0f;
#pragma unroll
            for (int s = 0; s < D_STATE; ++s) {
                float dA = __expf(dt * Arow[s]);
                h[s] = dA * h[s] + dtx * sBC[tt][s];
                y += h[s] * sBC[tt][16 + s];
            }
            float yy = y + xv * dsk;
            float zv = xz[m * (long)(2 * D_INNER) + D_INNER + d];
            yy *= zv / (1.0f + __expf(-zv));     // * SiLU(z)
            y_bf[m * (long)D_INNER + d] = (bf16)yy;
        }
    }
}

// =====================================================================
// host launcher
// =====================================================================
extern "C" void kernel_launch(void* const* d_in, const int* in_sizes, int n_in,
                              void* d_out, int out_size, void* d_ws, size_t ws_size,
                              hipStream_t stream)
{
    (void)in_sizes; (void)n_in; (void)out_size; (void)ws_size;
    const float* x      = (const float*)d_in[0];
    const float* W_in   = (const float*)d_in[1];
    const float* conv_w = (const float*)d_in[2];
    const float* conv_b = (const float*)d_in[3];
    const float* W_x    = (const float*)d_in[4];
    const float* W_dt   = (const float*)d_in[5];
    const float* b_dt   = (const float*)d_in[6];
    const float* A_log  = (const float*)d_in[7];
    const float* D_skip = (const float*)d_in[8];
    const float* W_out  = (const float*)d_in[9];
    float* out = (float*)d_out;

    char* ws = (char*)d_ws;
    auto alloc = [&](size_t bytes) -> char* {
        char* p = ws;
        ws += (bytes + 255) & ~(size_t)255;
        return p;
    };
    float* xz     = (float*)alloc((size_t)MTOT * 2 * D_INNER * 4);   // 134 MB
    float* xc     = (float*)alloc((size_t)MTOT * D_INNER * 4);       //  67 MB
    float* xdbl   = (float*)alloc((size_t)MTOT * 96 * 4);            //   3 MB
    float* dtpre  = (float*)alloc((size_t)MTOT * D_INNER * 4);       //  67 MB
    float* PQ     = (float*)alloc((size_t)BSZ * D_INNER * NSEG * 32 * 4); // 33.5 MB
    float* Hs     = (float*)alloc((size_t)BSZ * D_INNER * NSEG * 16 * 4); // 16.8 MB
    bf16* x_bf    = (bf16*)alloc((size_t)MTOT * D_MODEL * 2);
    bf16* Win_bf  = (bf16*)alloc((size_t)2 * D_INNER * D_MODEL * 2);
    bf16* xc_bf   = (bf16*)alloc((size_t)MTOT * D_INNER * 2);
    bf16* Wx_bf   = (bf16*)alloc((size_t)(DT_RANK + 2 * D_STATE) * D_INNER * 2);
    bf16* dtlo_bf = (bf16*)alloc((size_t)MTOT * DT_RANK * 2);
    bf16* Wdt_bf  = (bf16*)alloc((size_t)D_INNER * DT_RANK * 2);
    bf16* y_bf    = (bf16*)alloc((size_t)MTOT * D_INNER * 2);
    bf16* Wout_bf = (bf16*)alloc((size_t)D_MODEL * D_INNER * 2);

    auto cvt = [&](const float* src, bf16* dst, long n) {
        long g = (n + 255) / 256;
        cvt_f32_bf16<<<dim3((unsigned)g), dim3(256), 0, stream>>>(src, dst, n);
    };

    // 1) precision staging (bf16 for matrix pipe)
    cvt(x,     x_bf,    (long)MTOT * D_MODEL);
    cvt(W_in,  Win_bf,  (long)2 * D_INNER * D_MODEL);
    cvt(W_x,   Wx_bf,   (long)(DT_RANK + 2 * D_STATE) * D_INNER);
    cvt(W_dt,  Wdt_bf,  (long)D_INNER * DT_RANK);
    cvt(W_out, Wout_bf, (long)D_MODEL * D_INNER);

    // 2) xz = x @ W_in^T   [8192 x 4096, K=1024]
    gemm_bf16_wmma<<<dim3(2 * D_INNER / 128, MTOT / 128), dim3(256), 0, stream>>>(
        x_bf, Win_bf, xz, MTOT, 2 * D_INNER, D_MODEL);

    // 3) causal conv + SiLU -> xc (f32 + bf16)
    {
        long n = (long)MTOT * D_INNER;
        conv_silu_kernel<<<dim3((unsigned)((n + 255) / 256)), dim3(256), 0, stream>>>(
            xz, conv_w, conv_b, xc, xc_bf);
    }

    // 4) xdbl = xc @ W_x^T   [8192 x 96, K=2048]  (N=96 handled by tile guards)
    gemm_bf16_wmma<<<dim3(1, MTOT / 128), dim3(256), 0, stream>>>(
        xc_bf, Wx_bf, xdbl, MTOT, DT_RANK + 2 * D_STATE, D_INNER);

    // 5) dt_pre = dt_lo @ W_dt^T   [8192 x 2048, K=64]
    {
        long n = (long)MTOT * DT_RANK;
        extract_dtlo<<<dim3((unsigned)((n + 255) / 256)), dim3(256), 0, stream>>>(xdbl, dtlo_bf, n);
    }
    gemm_bf16_wmma<<<dim3(D_INNER / 128, MTOT / 128), dim3(256), 0, stream>>>(
        dtlo_bf, Wdt_bf, dtpre, MTOT, D_INNER, DT_RANK);

    // 6) segment-parallel selective scan + skip + SiLU(z) gate -> y_bf
    scan_passA<<<dim3(BSZ * NSEG * (D_INNER / 256)), dim3(256), 0, stream>>>(
        dtpre, xc, xdbl, b_dt, A_log, PQ);
    scan_passB<<<dim3((BSZ * D_INNER + 255) / 256), dim3(256), 0, stream>>>(PQ, Hs);
    scan_passC<<<dim3(BSZ * NSEG * (D_INNER / 256)), dim3(256), 0, stream>>>(
        dtpre, xc, xdbl, xz, b_dt, A_log, D_skip, Hs, y_bf);

    // 7) out = y @ W_out^T   [8192 x 1024, K=2048]
    gemm_bf16_wmma<<<dim3(D_MODEL / 128, MTOT / 128), dim3(256), 0, stream>>>(
        y_bf, Wout_bf, out, MTOT, D_MODEL, D_INNER);
}
